// LPGBlock_76982993813846
// MI455X (gfx1250) — matrix operand order repaired and checked by reading, not compiled, not run on hardware
//
#include <hip/hip_runtime.h>
#include <hip/hip_bf16.h>
#include <math.h>

typedef _Float16 v16h __attribute__((ext_vector_type(16)));
typedef _Float16 h8   __attribute__((ext_vector_type(8)));
typedef float    v8f  __attribute__((ext_vector_type(8)));
typedef float    f4   __attribute__((ext_vector_type(4)));

#define B_   16
#define C_   128
#define H_   44
#define W_   152
#define HW_  (H_*W_)          // 6688
#define HS_  (H_*8)           // 352
#define WS_  (W_*8)           // 1216
#define TILES_PER_B (HW_/16)  // 418
#define NTILES (B_*TILES_PER_B)
#define WAVES_PER_BLK 8
#define LANESTRIDE 24                  // halfs per lane slot in a weight fragment (48B aligned)
#define FRAGHALFS (32*LANESTRIDE)      // 768 halfs per fragment
#define NFRAG 23
#define ACTROW 72                      // halfs per pixel row in activation scratch (144B)

#define PI_F 3.1415926535f

__device__ __forceinline__ v8f wmma16(v16h a, v16h b, v8f c) {
    return __builtin_amdgcn_wmma_f32_16x16x32_f16(false, a, false, b, (short)0, c, false, false);
}

__device__ __forceinline__ float elu1(float z) {
    return z > 0.f ? z : (__expf(z) - 1.f);
}

__launch_bounds__(256)
__global__ void lpg_block_kernel(const float* __restrict__ x,
                                 const float* __restrict__ w0, const float* __restrict__ b0,
                                 const float* __restrict__ w1, const float* __restrict__ b1,
                                 const float* __restrict__ w2, const float* __restrict__ b2,
                                 const float* __restrict__ w3, const float* __restrict__ b3,
                                 const float* __restrict__ w4, const float* __restrict__ b4,
                                 const float* __restrict__ cw, const float* __restrict__ cb,
                                 float* __restrict__ out)
{
    __shared__ _Float16 sW[NFRAG*FRAGHALFS];           // weight fragments, B-layout
    __shared__ float    sBias[124];
    __shared__ float    sCw[12];
    __shared__ float    sCb[4];
    __shared__ _Float16 sAct[WAVES_PER_BLK*16*ACTROW]; // per-wave activation scratch
    __shared__ float    sA4[WAVES_PER_BLK*16*4];       // per-wave final activations

    const int tid = threadIdx.x;

    // ---------------- cooperative weight preload (f32 -> f16, B-fragment layout) ---------
    for (int idx = tid; idx < NFRAG*512; idx += 256) {
        int f = idx >> 9, r = idx & 511;
        int lane = r >> 4, e = r & 15;
        const float* wp; int ci, co, kcn, fb;
        if      (f < 16) { wp = w0; ci = 128; co = 64; kcn = 4; fb = 0;  }
        else if (f < 20) { wp = w1; ci = 64;  co = 32; kcn = 2; fb = 16; }
        else if (f < 21) { wp = w2; ci = 32;  co = 16; kcn = 1; fb = 20; }
        else if (f < 22) { wp = w3; ci = 16;  co = 8;  kcn = 1; fb = 21; }
        else             { wp = w4; ci = 8;   co = 4;  kcn = 1; fb = 22; }
        int fl = f - fb;
        int kc = fl % kcn, nt = fl / kcn;
        // B-matrix layout (32x16 f16): N=lane&15; elems 0..7 -> K=(lane>>4)*8+e, elems 8..15 -> K=16+(lane>>4)*8+(e-8)
        int o = nt*16 + (lane & 15);
        int k = kc*32 + (lane >> 4)*8 + (e & 7) + (e >> 3)*16;
        _Float16 v = (_Float16)0.f;
        if (o < co && k < ci) v = (_Float16)wp[(size_t)o*ci + k];
        sW[f*FRAGHALFS + lane*LANESTRIDE + e] = v;
    }
    if (tid < 124) {
        float v;
        if      (tid < 64)  v = b0[tid];
        else if (tid < 96)  v = b1[tid-64];
        else if (tid < 112) v = b2[tid-96];
        else if (tid < 120) v = b3[tid-112];
        else                v = b4[tid-120];
        sBias[tid] = v;
    }
    if (tid < 12) sCw[tid] = cw[tid];
    if (tid < 3)  sCb[tid] = cb[tid];
    __syncthreads();

    // ---------------- per-wave tile ----------------
    const int lane   = tid & 31;
    const int waveId = tid >> 5;
    const int tile   = blockIdx.x * WAVES_PER_BLK + waveId;
    if (tile >= NTILES) return;

    const int m  = lane & 15;   // A-layout: pixel row; D-layout: channel column
    const int hi = lane >> 4;
    const int b  = tile / TILES_PER_B;
    const int p0 = (tile - b*TILES_PER_B) * 16;  // hw index of first pixel (tiles never straddle b)
    const int hw = p0 + m;
    const float* px = x + (size_t)b*C_*HW_ + hw;

    _Float16* act = &sAct[waveId*16*ACTROW];

    // B fragment from LDS (two 16B ds loads per lane)
    auto loadB = [&](int f) -> v16h {
        const _Float16* p = &sW[f*FRAGHALFS + lane*LANESTRIDE];
        h8 lo = *(const h8*)p;
        h8 hv = *(const h8*)(p + 8);
        return __builtin_shufflevector(lo, hv, 0,1,2,3,4,5,6,7,8,9,10,11,12,13,14,15);
    };
    // A fragment from activation scratch (A-layout: elems 0..7 = K(kbase+hi*8+e), 8..15 = +16)
    auto loadA = [&](int kbase, bool hiZero) -> v16h {
        const _Float16* p = &act[m*ACTROW + kbase + hi*8];
        h8 lo = *(const h8*)p;
        h8 hv;
        if (hiZero) {
            #pragma unroll
            for (int i = 0; i < 8; ++i) hv[i] = (_Float16)0.f;
        } else {
            hv = *(const h8*)(p + 16);
        }
        return __builtin_shufflevector(lo, hv, 0,1,2,3,4,5,6,7,8,9,10,11,12,13,14,15);
    };
    // bias + ELU + f16 store to activation scratch (D-layout -> [pixel][channel])
    auto storeTile = [&](v8f acc, int nt, int bbase, int nvalid) {
        int n = m;  // D-layout: channel = lane&15
        if (n < nvalid) {
            float bias = sBias[bbase + nt*16 + n];
            #pragma unroll
            for (int v = 0; v < 8; ++v) {
                int mm = v + 8*hi;  // D-layout: pixel = vgpr + 8*(lane>=16)
                act[mm*ACTROW + nt*16 + n] = (_Float16)elu1(acc[v] + bias);
            }
        }
    };

    // ---- layer 0: 128 -> 64 (A gathered from global, f32 -> f16) ----
    v16h A0[4];
    #pragma unroll
    for (int kc = 0; kc < 4; ++kc) {
        int kb = kc*32 + hi*8;
        #pragma unroll
        for (int e = 0; e < 8; ++e) {
            A0[kc][e]     = (_Float16)px[(size_t)(kb + e)      * HW_];
            A0[kc][e + 8] = (_Float16)px[(size_t)(kb + 16 + e) * HW_];
        }
    }
    v8f acc0[4];
    #pragma unroll
    for (int nt = 0; nt < 4; ++nt) {
        #pragma unroll
        for (int i = 0; i < 8; ++i) acc0[nt][i] = 0.f;
        #pragma unroll
        for (int kc = 0; kc < 4; ++kc)
            acc0[nt] = wmma16(A0[kc], loadB(nt*4 + kc), acc0[nt]);
    }
    #pragma unroll
    for (int nt = 0; nt < 4; ++nt) storeTile(acc0[nt], nt, 0, 16);

    // ---- layer 1: 64 -> 32 ----
    v16h A1[2] = { loadA(0, false), loadA(32, false) };
    v8f acc1[2];
    #pragma unroll
    for (int nt = 0; nt < 2; ++nt) {
        #pragma unroll
        for (int i = 0; i < 8; ++i) acc1[nt][i] = 0.f;
        #pragma unroll
        for (int kc = 0; kc < 2; ++kc)
            acc1[nt] = wmma16(A1[kc], loadB(16 + nt*2 + kc), acc1[nt]);
    }
    storeTile(acc1[0], 0, 64, 16);
    storeTile(acc1[1], 1, 64, 16);

    // ---- layer 2: 32 -> 16 ----
    {
        v16h A2 = loadA(0, false);
        v8f acc2;
        #pragma unroll
        for (int i = 0; i < 8; ++i) acc2[i] = 0.f;
        acc2 = wmma16(A2, loadB(20), acc2);
        storeTile(acc2, 0, 96, 16);
    }

    // ---- layer 3: 16 -> 8 (K padded to 32) ----
    {
        v16h A3 = loadA(0, true);   // valid K0..15, high half zero
        v8f acc3;
        #pragma unroll
        for (int i = 0; i < 8; ++i) acc3[i] = 0.f;
        acc3 = wmma16(A3, loadB(21), acc3);
        storeTile(acc3, 0, 112, 8);
    }

    // ---- layer 4: 8 -> 4 (K padded to 32, only lanes<16 carry data) ----
    {
        v16h A4v;
        #pragma unroll
        for (int i = 0; i < 16; ++i) A4v[i] = (_Float16)0.f;
        if (hi == 0) {
            h8 lo = *(const h8*)&act[m*ACTROW];   // K0..7
            #pragma unroll
            for (int i = 0; i < 8; ++i) A4v[i] = lo[i];
        }
        v8f acc4;
        #pragma unroll
        for (int i = 0; i < 8; ++i) acc4[i] = 0.f;
        acc4 = wmma16(A4v, loadB(22), acc4);
        int n = m;
        if (n < 4) {
            float bias = sBias[120 + n];
            #pragma unroll
            for (int v = 0; v < 8; ++v) {
                int mm = v + 8*hi;
                sA4[(waveId*16 + mm)*4 + n] = elu1(acc4[v] + bias);
            }
        }
    }

    // ---- head + plane-to-depth expansion: 2 lanes per pixel, 4 sv rows each ----
    {
        int pp  = lane >> 1;
        int svh = (lane & 1) * 4;
        const float* a = &sA4[(waveId*16 + pp)*4];
        float a0v = a[0], a1v = a[1], a2v = a[2], a3v = a[3];
        float y0 = sCb[0] + sCw[0]*a0v + sCw[1]*a1v + sCw[2]*a2v  + sCw[3]*a3v;
        float y1 = sCb[1] + sCw[4]*a0v + sCw[5]*a1v + sCw[6]*a2v  + sCw[7]*a3v;
        float y2 = sCb[2] + sCw[8]*a0v + sCw[9]*a1v + sCw[10]*a2v + sCw[11]*a3v;

        float theta = (1.f/(1.f+__expf(-y0))) * (PI_F/6.f);
        float phi   = (1.f/(1.f+__expf(-y1))) * (PI_F*2.f);
        float dist  = (1.f/(1.f+__expf(-y2))) * 83.f;

        float st, ct, sp, cp;
        __sincosf(theta, &st, &ct);
        __sincosf(phi,   &sp, &cp);
        float n1 = st*cp, n2 = st*sp, n3 = ct;
        float inv = 1.f / fmaxf(sqrtf(n1*n1 + n2*n2 + n3*n3), 1e-12f);
        n1 *= inv; n2 *= inv; n3 *= inv;

        int P = p0 + pp;
        int h = P / W_, w = P - h*W_;
        float* orow = out + ((size_t)b*HS_ + (size_t)h*8) * WS_ + (size_t)w*8;
        const float sc = 1.f / (8.f * 715.f);

        #pragma unroll
        for (int jv = 0; jv < 4; ++jv) {
            int   sv   = svh + jv;
            float voff = ((float)sv - 3.5f) * sc;
            float base = n2*voff + n3;
            f4 d0, d1;
            #pragma unroll
            for (int i = 0; i < 4; ++i)
                d0[i] = dist / (n1 * (((float)i - 3.5f) * sc) + base);
            #pragma unroll
            for (int i = 0; i < 4; ++i)
                d1[i] = dist / (n1 * (((float)(i + 4) - 3.5f) * sc) + base);
            float* dst = orow + (size_t)sv * WS_;
            *(f4*)dst       = d0;
            *(f4*)(dst + 4) = d1;
        }
    }
}

extern "C" void kernel_launch(void* const* d_in, const int* in_sizes, int n_in,
                              void* d_out, int out_size, void* d_ws, size_t ws_size,
                              hipStream_t stream) {
    const float* x  = (const float*)d_in[0];
    const float* w0 = (const float*)d_in[1];
    const float* b0 = (const float*)d_in[2];
    const float* w1 = (const float*)d_in[3];
    const float* b1 = (const float*)d_in[4];
    const float* w2 = (const float*)d_in[5];
    const float* b2 = (const float*)d_in[6];
    const float* w3 = (const float*)d_in[7];
    const float* b3 = (const float*)d_in[8];
    const float* w4 = (const float*)d_in[9];
    const float* b4 = (const float*)d_in[10];
    const float* cw = (const float*)d_in[11];
    const float* cb = (const float*)d_in[12];
    float* out = (float*)d_out;

    dim3 grid((NTILES + WAVES_PER_BLK - 1) / WAVES_PER_BLK);  // 836 blocks x 8 waves = 6688 tiles
    lpg_block_kernel<<<grid, 256, 0, stream>>>(x, w0, b0, w1, b1, w2, b2, w3, b3, w4, b4, cw, cb, out);
}
